// Informer_29566554866206
// MI455X (gfx1250) — matrix operand order repaired and checked by reading, not compiled, not run on hardware
//
#include <hip/hip_runtime.h>
#include <hip/hip_bf16.h>

// ---------------- problem constants ----------------
constexpr int Bc  = 2;
constexpr int Lc  = 4096;
constexpr int DMc = 512;
constexpr int Hc  = 8;
constexpr int DFc = 512;
constexpr int Dc  = 64;          // DM / H
constexpr int Uc  = 45;          // min(C*ceil(ln L), L) = 5*9
constexpr float SCALE_C = 0.125f; // 1/sqrt(64)
constexpr float EPS_C   = 0.001f;

typedef __attribute__((ext_vector_type(16))) __bf16 v16bf;
typedef __attribute__((ext_vector_type(8)))  __bf16 v8bf;
typedef __attribute__((ext_vector_type(8)))  float  v8f;

// ---------------- CDNA5 async global->LDS staging ----------------
#if __has_builtin(__builtin_amdgcn_global_load_async_to_lds_b128)
  #define ASYNC_TO_LDS_B128 __builtin_amdgcn_global_load_async_to_lds_b128
  #define HAVE_ASYNC_LDS 1
#elif __has_builtin(__builtin_amdgcn_global_async_load_to_lds_b128)
  #define ASYNC_TO_LDS_B128 __builtin_amdgcn_global_async_load_to_lds_b128
  #define HAVE_ASYNC_LDS 1
#else
  #define HAVE_ASYNC_LDS 0
#endif

// 16-byte int vector, matching the builtin's parameter type from the diagnostic:
// '__attribute__((__vector_size__(4 * sizeof(int)))) int __device__ *'
typedef int vi4 __attribute__((vector_size(4 * sizeof(int))));
typedef __attribute__((address_space(1))) vi4* as1_vi4;   // global
typedef __attribute__((address_space(3))) vi4* as3_vi4;   // LDS

__device__ __forceinline__ void async_copy16(const __bf16* g, __bf16* l) {
#if HAVE_ASYNC_LDS
  ASYNC_TO_LDS_B128((as1_vi4)(void*)const_cast<__bf16*>(g),
                    (as3_vi4)(void*)l,
                    0 /*imm offset*/, 0 /*cpol*/);
#else
  *(uint4*)l = *(const uint4*)g;
#endif
}

__device__ __forceinline__ void wait_tiles() {
#if HAVE_ASYNC_LDS
  #if __has_builtin(__builtin_amdgcn_s_wait_asynccnt)
  __builtin_amdgcn_s_wait_asynccnt(0);
  #else
  asm volatile("s_wait_asynccnt 0x0" ::: "memory");
  #endif
#endif
  __syncthreads();
}

// ---------------- fp32 -> bf16 conversion ----------------
__global__ void cvt_bf16(const float* __restrict__ in, __bf16* __restrict__ out, int n) {
  int i = blockIdx.x * blockDim.x + threadIdx.x;
  if (i < n) out[i] = (__bf16)in[i];
}

// W[K,N] fp32 (row-major) -> Wt[N,K] bf16 (each output row = one column of W)
__global__ void cvt_bf16_t(const float* __restrict__ in, __bf16* __restrict__ out,
                           int K, int N) {
  int i = blockIdx.x * blockDim.x + threadIdx.x;
  if (i < K * N) {
    int k = i / N, n = i % N;
    out[(size_t)n * K + k] = (__bf16)in[i];
  }
}

// ---------------- WMMA bf16 GEMM, LDS double-buffered ----------------
// C[M,N] = A[M,K] (bf16, row-major) @ Wt[N,K]^T (bf16) + bias[N], opt relu.
// block = 256 threads (8 waves), tile 128x128, k-step 32.
// Wave (wm=wave>>1, wn=wave&1) computes rows [wm*32,+32) x cols [wn*64,+64):
// 2x4 accumulators of 16x16.
constexpr int BM = 128, BN = 128, BK = 32;
constexpr int SA = 56;   // LDS row stride in elems: 112 B (16B-aligned, 28-bank stride)

template <bool RELU>
__global__ __launch_bounds__(256)
void gemm_wmma_bf16(const __bf16* __restrict__ A,
                    const __bf16* __restrict__ Wt,
                    const float* __restrict__ bias,
                    float* __restrict__ C,
                    int M, int N, int K) {
  __shared__ __bf16 As[2][BM * SA];
  __shared__ __bf16 Bs[2][BN * SA];

  const int tid  = threadIdx.x;
  const int lane = tid & 31;
  const int wave = tid >> 5;   // 0..7
  const int wm   = wave >> 1;  // 0..3
  const int wn   = wave & 1;   // 0..1
  const int m0   = blockIdx.x * BM;
  const int n0   = blockIdx.y * BN;
  const int m16  = lane & 15;
  const int half = lane >> 4;  // 0 or 1

  v8f acc[2][4];
#pragma unroll
  for (int i = 0; i < 2; ++i)
#pragma unroll
    for (int j = 0; j < 4; ++j) acc[i][j] = v8f{};

  // 128 rows x 32 elems per matrix per k-step = 512 16B-chunks; 2 per thread.
  auto copy_tiles = [&](int k0, int bufi) {
#pragma unroll
    for (int i = 0; i < 2; ++i) {
      int e   = tid + i * 256;
      int row = e >> 2;
      int c0  = (e & 3) * 8;
      async_copy16(A  + (size_t)(m0 + row) * K + k0 + c0, &As[bufi][row * SA + c0]);
      async_copy16(Wt + (size_t)(n0 + row) * K + k0 + c0, &Bs[bufi][row * SA + c0]);
    }
  };

  int buf = 0;
  copy_tiles(0, 0);

  for (int k0 = 0; k0 < K; k0 += BK) {
    wait_tiles();                              // current buffer resident, all waves past prev reads
    if (k0 + BK < K) copy_tiles(k0 + BK, buf ^ 1);   // stream next tile while computing

    // A fragment (16x32 bf16): elems 0..7 -> k = half*8 + j ; 8..15 -> k = 16 + half*8 + j
    v16bf afrag[2];
#pragma unroll
    for (int t = 0; t < 2; ++t) {
      const __bf16* p = &As[buf][(wm * 32 + t * 16 + m16) * SA];
      v8bf lo = *(const v8bf*)(p + half * 8);
      v8bf hi = *(const v8bf*)(p + 16 + half * 8);
#pragma unroll
      for (int j = 0; j < 8; ++j) { afrag[t][j] = lo[j]; afrag[t][8 + j] = hi[j]; }
    }
    // B fragment (32x16 bf16): elems 0..7 -> k = half*8 + j ; 8..15 -> k = 16 + half*8 + j
    v16bf bfrag[4];
#pragma unroll
    for (int t = 0; t < 4; ++t) {
      const __bf16* p = &Bs[buf][(wn * 64 + t * 16 + m16) * SA];
      v8bf lo = *(const v8bf*)(p + half * 8);
      v8bf hi = *(const v8bf*)(p + 16 + half * 8);
#pragma unroll
      for (int j = 0; j < 8; ++j) { bfrag[t][j] = lo[j]; bfrag[t][8 + j] = hi[j]; }
    }
#pragma unroll
    for (int i = 0; i < 2; ++i)
#pragma unroll
      for (int j = 0; j < 4; ++j)
        acc[i][j] = __builtin_amdgcn_wmma_f32_16x16x32_bf16(
            false, afrag[i], false, bfrag[j], (short)0, acc[i][j], false, false);
    buf ^= 1;
  }

  // C/D layout: VGPR r -> m = base + half*8 + r, n = base + (lane&15)
  const int nbase = n0 + wn * 64;
#pragma unroll
  for (int j = 0; j < 4; ++j) {
    const int nn = nbase + j * 16 + m16;
    const float bn = bias ? bias[nn] : 0.0f;
#pragma unroll
    for (int i = 0; i < 2; ++i) {
#pragma unroll
      for (int r = 0; r < 8; ++r) {
        int m = m0 + wm * 32 + i * 16 + half * 8 + r;
        float val = acc[i][j][r] + bn;
        if (RELU) val = fmaxf(val, 0.0f);
        C[(size_t)m * N + nn] = val;
      }
    }
  }
}

// ---------------- sampled-score sparsity metric ----------------
__global__ void sampled_metric(const float* __restrict__ q, const float* __restrict__ k,
                               const int* __restrict__ idx, float* __restrict__ Mout) {
  int wid  = (blockIdx.x * blockDim.x + threadIdx.x) >> 5;   // 0 .. B*H*L-1
  int lane = threadIdx.x & 31;
  int l  = wid % Lc;
  int bh = wid / Lc;
  int h  = bh % Hc;
  int b  = bh / Hc;
  const float* qrow = q + (size_t)(b * Lc + l) * DMc + h * Dc;

  float mx = -3.0e38f, sm = 0.0f;
  for (int u = lane; u < Uc; u += 32) {
    int j = idx[(size_t)l * Uc + u];
    const float* krow = k + (size_t)(b * Lc + j) * DMc + h * Dc;
    float dot = 0.0f;
#pragma unroll
    for (int d = 0; d < Dc; ++d) dot += qrow[d] * krow[d];
    mx = fmaxf(mx, dot);
    sm += dot;
  }
  for (int off = 16; off; off >>= 1) {
    mx = fmaxf(mx, __shfl_xor(mx, off, 32));
    sm += __shfl_xor(sm, off, 32);
  }
  if (lane == 0) Mout[(size_t)bh * Lc + l] = mx - sm / (float)Uc;
}

// ---------------- top-k (iterative argmax, ties -> lower index) ----------------
__global__ void topk_iter(float* __restrict__ Mbuf, int* __restrict__ topidx) {
  int bh = blockIdx.x;
  float* mrow = Mbuf + (size_t)bh * Lc;
  __shared__ float sv[256];
  __shared__ int   si[256];
  for (int it = 0; it < Uc; ++it) {
    float bv = -3.0e38f; int bi = 0;
    for (int l = threadIdx.x; l < Lc; l += 256) {
      float v = mrow[l];
      if (v > bv) { bv = v; bi = l; }
    }
    sv[threadIdx.x] = bv; si[threadIdx.x] = bi;
    __syncthreads();
    for (int s = 128; s; s >>= 1) {
      if (threadIdx.x < s) {
        if (sv[threadIdx.x + s] > sv[threadIdx.x] ||
            (sv[threadIdx.x + s] == sv[threadIdx.x] && si[threadIdx.x + s] < si[threadIdx.x])) {
          sv[threadIdx.x] = sv[threadIdx.x + s];
          si[threadIdx.x] = si[threadIdx.x + s];
        }
      }
      __syncthreads();
    }
    if (threadIdx.x == 0) { topidx[bh * Uc + it] = si[0]; mrow[si[0]] = -3.0e38f; }
    __syncthreads();
  }
}

// ---------------- dense attention for selected queries ----------------
__global__ void sparse_attn(const float* __restrict__ q, const float* __restrict__ k,
                            const float* __restrict__ v, const int* __restrict__ topidx,
                            float* __restrict__ outtop) {
  __shared__ float attn[Lc];    // 16 KB
  __shared__ float sq[Dc];
  __shared__ float red[256];
  int bh = blockIdx.x / Uc, u = blockIdx.x % Uc;
  int b = bh / Hc, h = bh % Hc;
  int lq = topidx[bh * Uc + u];
  if (threadIdx.x < Dc)
    sq[threadIdx.x] = q[(size_t)(b * Lc + lq) * DMc + h * Dc + threadIdx.x];
  __syncthreads();

  float lmax = -3.0e38f;
  for (int l = threadIdx.x; l < Lc; l += 256) {
    const float* krow = k + (size_t)(b * Lc + l) * DMc + h * Dc;
    float dot = 0.0f;
#pragma unroll
    for (int d = 0; d < Dc; ++d) dot += sq[d] * krow[d];
    dot *= SCALE_C;
    attn[l] = dot;
    lmax = fmaxf(lmax, dot);
  }
  red[threadIdx.x] = lmax; __syncthreads();
  for (int s = 128; s; s >>= 1) {
    if (threadIdx.x < s) red[threadIdx.x] = fmaxf(red[threadIdx.x], red[threadIdx.x + s]);
    __syncthreads();
  }
  float mx = red[0]; __syncthreads();

  float lsum = 0.0f;
  for (int l = threadIdx.x; l < Lc; l += 256) {
    float e = __expf(attn[l] - mx);
    attn[l] = e;
    lsum += e;
  }
  red[threadIdx.x] = lsum; __syncthreads();
  for (int s = 128; s; s >>= 1) {
    if (threadIdx.x < s) red[threadIdx.x] += red[threadIdx.x + s];
    __syncthreads();
  }
  float inv = 1.0f / red[0];
  __syncthreads();

  if (threadIdx.x < Dc) {
    int d = threadIdx.x;
    float acc = 0.0f;
    for (int l = 0; l < Lc; ++l)
      acc += attn[l] * v[(size_t)(b * Lc + l) * DMc + h * Dc + d];
    outtop[(size_t)(bh * Uc + u) * Dc + d] = acc * inv;
  }
}

// ---------------- V mean over sequence ----------------
__global__ void mean_v(const float* __restrict__ v, float* __restrict__ meanv) {
  int i = blockIdx.x * blockDim.x + threadIdx.x;   // B*H*D = 1024
  if (i >= Bc * Hc * Dc) return;
  int d = i % Dc; int h = (i / Dc) % Hc; int b = i / (Dc * Hc);
  float s = 0.0f;
  for (int l = 0; l < Lc; ++l)
    s += v[(size_t)(b * Lc + l) * DMc + h * Dc + d];
  meanv[i] = s / (float)Lc;
}

// ctx = broadcast(mean_v) reshaped to (B, L, DM)
__global__ void ctx_broadcast(const float* __restrict__ meanv, float* __restrict__ t0) {
  size_t i = (size_t)blockIdx.x * blockDim.x + threadIdx.x;   // B*L*DM
  if (i >= (size_t)Bc * Lc * DMc) return;
  int c = (int)(i % DMc);
  int b = (int)(i / ((size_t)DMc * Lc));
  int h = c / Dc, d = c % Dc;
  t0[i] = meanv[(b * Hc + h) * Dc + d];
}

// scatter attention outputs to their rows
__global__ void ctx_scatter(const float* __restrict__ outtop, const int* __restrict__ topidx,
                            float* __restrict__ t0) {
  int i = blockIdx.x * blockDim.x + threadIdx.x;   // B*H*U*D
  if (i >= Bc * Hc * Uc * Dc) return;
  int d = i % Dc; int u = (i / Dc) % Uc; int bh = i / (Dc * Uc);
  int b = bh / Hc, h = bh % Hc;
  int l = topidx[bh * Uc + u];
  t0[(size_t)(b * Lc + l) * DMc + h * Dc + d] = outtop[i];
}

// ---------------- residual add + LayerNorm (one block per row) ----------------
__global__ void add_layernorm(const float* __restrict__ x, const float* __restrict__ y,
                              const float* __restrict__ g, const float* __restrict__ beta,
                              float* __restrict__ out) {
  __shared__ float row[DMc];
  __shared__ float red[256];
  size_t r = blockIdx.x;
  const float* xr = x + r * DMc;
  const float* yr = y + r * DMc;
  float s = 0.0f;
  for (int i = threadIdx.x; i < DMc; i += 256) {
    float t = xr[i] + yr[i];
    row[i] = t;
    s += t;
  }
  red[threadIdx.x] = s; __syncthreads();
  for (int st = 128; st; st >>= 1) {
    if (threadIdx.x < st) red[threadIdx.x] += red[threadIdx.x + st];
    __syncthreads();
  }
  float mu = red[0] / (float)DMc; __syncthreads();
  float vs = 0.0f;
  for (int i = threadIdx.x; i < DMc; i += 256) {
    float t = row[i] - mu;
    vs += t * t;
  }
  red[threadIdx.x] = vs; __syncthreads();
  for (int st = 128; st; st >>= 1) {
    if (threadIdx.x < st) red[threadIdx.x] += red[threadIdx.x + st];
    __syncthreads();
  }
  float inv = rsqrtf(red[0] / (float)DMc + EPS_C);
  for (int i = threadIdx.x; i < DMc; i += 256)
    out[r * DMc + i] = (row[i] - mu) * inv * g[i] + beta[i];
}

// ---------------- host orchestration ----------------
extern "C" void kernel_launch(void* const* d_in, const int* in_sizes, int n_in,
                              void* d_out, int out_size, void* d_ws, size_t ws_size,
                              hipStream_t stream) {
  (void)in_sizes; (void)n_in; (void)out_size; (void)ws_size;
  const float* x     = (const float*)d_in[0];
  const float* Wq    = (const float*)d_in[1];
  const float* bq    = (const float*)d_in[2];
  const float* Wk    = (const float*)d_in[3];
  const float* bk    = (const float*)d_in[4];
  const float* Wv    = (const float*)d_in[5];
  const float* bv    = (const float*)d_in[6];
  const float* Wo    = (const float*)d_in[7];
  const float* bo    = (const float*)d_in[8];
  const float* W1    = (const float*)d_in[9];
  const float* b1    = (const float*)d_in[10];
  const float* W2    = (const float*)d_in[11];
  const float* b2    = (const float*)d_in[12];
  const float* g1    = (const float*)d_in[13];
  const float* beta1 = (const float*)d_in[14];
  const float* g2    = (const float*)d_in[15];
  const float* beta2 = (const float*)d_in[16];
  const int*   idx   = (const int*)d_in[17];
  float* out = (float*)d_out;

  char* ws = (char*)d_ws;
  auto alloc = [&](size_t bytes) -> char* {
    char* p = ws;
    ws += (bytes + 255) & ~(size_t)255;
    return p;
  };

  const size_t MN = (size_t)Bc * Lc * DMc;            // 4,194,304 elems
  float*  qf     = (float*)alloc(MN * 4);
  float*  kf     = (float*)alloc(MN * 4);
  float*  vf     = (float*)alloc(MN * 4);
  float*  x1f    = (float*)alloc(MN * 4);
  float*  t2     = (float*)alloc(MN * 4);
  __bf16* xb     = (__bf16*)alloc(MN * 2);
  __bf16* t0b    = (__bf16*)alloc(MN * 2);
  __bf16* wb     = (__bf16*)alloc((size_t)DMc * DMc * 2);
  float*  Mbuf   = (float*)alloc((size_t)Bc * Hc * Lc * 4);
  int*    topidx = (int*)alloc((size_t)Bc * Hc * Uc * 4);
  float*  outtop = (float*)alloc((size_t)Bc * Hc * Uc * Dc * 4);
  float*  meanv  = (float*)alloc((size_t)Bc * Hc * Dc * 4);
  float*  t0 = qf;   // attn-in, reuses q after attention is done
  float*  t1 = kf;   // attn-out / ffn hidden, reuses k after attention is done

  const int M = Bc * Lc;                  // 8192
  const dim3 ggrid(M / BM, DMc / BN);     // (64, 4)
  const int CB = 256;
  const int nMN = (int)MN;
  const int nW  = DMc * DMc;

  // x -> bf16
  cvt_bf16<<<(nMN + CB - 1) / CB, CB, 0, stream>>>(x, xb, nMN);

  // Q, K, V projections
  cvt_bf16_t<<<(nW + CB - 1) / CB, CB, 0, stream>>>(Wq, wb, DMc, DMc);
  gemm_wmma_bf16<false><<<ggrid, 256, 0, stream>>>(xb, wb, bq, qf, M, DMc, DMc);
  cvt_bf16_t<<<(nW + CB - 1) / CB, CB, 0, stream>>>(Wk, wb, DMc, DMc);
  gemm_wmma_bf16<false><<<ggrid, 256, 0, stream>>>(xb, wb, bk, kf, M, DMc, DMc);
  cvt_bf16_t<<<(nW + CB - 1) / CB, CB, 0, stream>>>(Wv, wb, DMc, DMc);
  gemm_wmma_bf16<false><<<ggrid, 256, 0, stream>>>(xb, wb, bv, vf, M, DMc, DMc);

  // sparsity metric + top-k
  sampled_metric<<<(Bc * Hc * Lc) / 8, CB, 0, stream>>>(qf, kf, idx, Mbuf);
  topk_iter<<<Bc * Hc, CB, 0, stream>>>(Mbuf, topidx);

  // attention for selected queries; mean context elsewhere
  sparse_attn<<<Bc * Hc * Uc, CB, 0, stream>>>(qf, kf, vf, topidx, outtop);
  mean_v<<<(Bc * Hc * Dc + CB - 1) / CB, CB, 0, stream>>>(vf, meanv);
  ctx_broadcast<<<(nMN + CB - 1) / CB, CB, 0, stream>>>(meanv, t0);
  ctx_scatter<<<(Bc * Hc * Uc * Dc + CB - 1) / CB, CB, 0, stream>>>(outtop, topidx, t0);

  // output projection + residual LN
  cvt_bf16<<<(nMN + CB - 1) / CB, CB, 0, stream>>>(t0, t0b, nMN);
  cvt_bf16_t<<<(nW + CB - 1) / CB, CB, 0, stream>>>(Wo, wb, DMc, DMc);
  gemm_wmma_bf16<false><<<ggrid, 256, 0, stream>>>(t0b, wb, bo, t1, M, DMc, DMc);
  add_layernorm<<<Bc * Lc, CB, 0, stream>>>(x, t1, g1, beta1, x1f);

  // FFN
  cvt_bf16<<<(nMN + CB - 1) / CB, CB, 0, stream>>>(x1f, t0b, nMN);
  cvt_bf16_t<<<(nW + CB - 1) / CB, CB, 0, stream>>>(W1, wb, DMc, DFc);
  gemm_wmma_bf16<true><<<ggrid, 256, 0, stream>>>(t0b, wb, b1, t1, M, DFc, DMc);
  cvt_bf16<<<(nMN + CB - 1) / CB, CB, 0, stream>>>(t1, t0b, nMN);
  cvt_bf16_t<<<(nW + CB - 1) / CB, CB, 0, stream>>>(W2, wb, DFc, DMc);
  gemm_wmma_bf16<false><<<ggrid, 256, 0, stream>>>(t0b, wb, b2, t2, M, DMc, DFc);
  add_layernorm<<<Bc * Lc, CB, 0, stream>>>(x1f, t2, g2, beta2, out);
}
